// DeepseekDecoderLayer_70635032150794
// MI455X (gfx1250) — matrix operand order repaired and checked by reading, not compiled, not run on hardware
//
#include <hip/hip_runtime.h>

// ---------------------------------------------------------------------------
// DeepSeek-style decoder layer for MI455X (gfx1250, wave32, WMMA bf16).
// B=1, S=512, D=2048, H=KVH=16, HD=128, E=64, topK=6, I=1408, ISH=2816.
// Memory-bound on fp32 expert weights (~2.2 GB). Weights are async-DMA'd
// into LDS (global_load_async_to_lds_b128, ASYNCcnt double-buffer, staged
// two k-steps ahead), converted with v_cvt_pk_bf16_f32, and consumed by
// v_wmma_f32_16x16x32_bf16 with fp32 accumulation.
// Routed experts are gathered (3072 token-expert rows) into grouped GEMMs.
// Wave tile = 32x64 (2 A-frags x 4 B-frags -> 8 WMMA).
// ---------------------------------------------------------------------------

typedef __attribute__((ext_vector_type(16))) __bf16 v16bf;
typedef __attribute__((ext_vector_type(8)))  __bf16 v8bf;
typedef __attribute__((ext_vector_type(2)))  __bf16 v2bf;
typedef __attribute__((ext_vector_type(8)))  float  v8f;

#define TOK 512
#define DIM 2048
#define NH 16
#define HDIM 128
#define QKVN 6144
#define NEXP 64
#define TOPK 6
#define IMOE 1408
#define ISHARE 2816
#define ATTN_SCALE 0.08838834764831845f  // 128^-0.5

#if defined(__has_builtin)
#if __has_builtin(__builtin_amdgcn_cvt_pk_bf16_f32)
#define HAVE_CVT_PK_BF16 1
#endif
#if __has_builtin(__builtin_amdgcn_s_wait_asynccnt)
#define WAIT_ASYNC(n) __builtin_amdgcn_s_wait_asynccnt((unsigned short)(n))
#endif
#endif
#ifndef WAIT_ASYNC
#define WAIT_ASYNC(n) asm volatile("s_wait_asynccnt %0" ::"i"(n) : "memory")
#endif

// Async global -> LDS 16B copy (per lane), tracked by ASYNCcnt. GVS mode:
// addr = saddr(base) + vaddr(32-bit byte offset).
#define ASYNC_LD16(ldsoff, voff, base)                                        \
  asm volatile("global_load_async_to_lds_b128 %0, %1, %2"                     \
               ::"v"(ldsoff), "v"(voff), "s"(base) : "memory")

__device__ __forceinline__ __bf16 f2bf(float f) { return (__bf16)f; }

// Convert a pair of floats into r[idx], r[idx+1] using v_cvt_pk_bf16_f32.
__device__ __forceinline__ void cvt2(v16bf& r, int idx, float x, float y) {
#ifdef HAVE_CVT_PK_BF16
  auto pk = __builtin_amdgcn_cvt_pk_bf16_f32(x, y);
  v2bf v; __builtin_memcpy(&v, &pk, sizeof(v));
  r[idx] = v[0]; r[idx + 1] = v[1];
#else
  r[idx] = (__bf16)x; r[idx + 1] = (__bf16)y;
#endif
}

__device__ __forceinline__ v8f wmma_bf16(v16bf a, v16bf b, v8f c) {
  return __builtin_amdgcn_wmma_f32_16x16x32_bf16(false, a, false, b, (short)0, c,
                                                 false, false);
}

// A fragment (16x32 bf16), row-major bf16 source. arow already includes lane's m.
__device__ __forceinline__ v16bf a_frag_bf16(const __bf16* A, long arow, long lda,
                                             int kbase, int lane) {
  int half = (lane >> 4) & 1;
  const __bf16* base = A + arow * lda + kbase;
  v8bf a0 = *(const v8bf*)(base + half * 16);
  v8bf a1 = *(const v8bf*)(base + half * 16 + 8);
  v16bf r;
#pragma unroll
  for (int i = 0; i < 8; i++) { r[i] = a0[i]; r[8 + i] = a1[i]; }
  return r;
}

// A fragment (16x32) from fp32 row-major source, convert to bf16.
__device__ __forceinline__ v16bf a_frag_f32(const float* A, long arow, long lda,
                                            int kbase, int lane) {
  int half = (lane >> 4) & 1;
  const float* base = A + arow * lda + kbase;
  const float4* p0 = (const float4*)(base + half * 16);
  const float4* p1 = (const float4*)(base + half * 16 + 8);
  float4 f[4] = { p0[0], p0[1], p1[0], p1[1] };
  v16bf r;
#pragma unroll
  for (int c = 0; c < 4; c++) {
    cvt2(r, c * 4 + 0, f[c].x, f[c].y);
    cvt2(r, c * 4 + 2, f[c].z, f[c].w);
  }
  return r;
}

// B fragment (32x16) from a [64 x 32] fp32 tile in LDS (weight strip slice).
// B-column n = tile row (srow+n); k contiguous.
__device__ __forceinline__ v16bf b_frag_lds_f32(const float* T, int srow, int lane) {
  int n = lane & 15, half = (lane >> 4) & 1;
  const float4* p = (const float4*)(T + (srow + n) * 32 + half * 16);
  v16bf r;
#pragma unroll
  for (int c = 0; c < 4; c++) {
    float4 f = p[c];
    cvt2(r, c * 4 + 0, f.x, f.y);
    cvt2(r, c * 4 + 2, f.z, f.w);
  }
  return r;
}

// B fragment (32x16) from global, column n = row (nbase+n) of row-major W[N][K].
__device__ __forceinline__ v16bf b_frag_f32(const float* W, int nbase, long ldw,
                                            int kbase, int lane) {
  int n = lane & 15, half = (lane >> 4) & 1;
  const float4* p = (const float4*)(W + (long)(nbase + n) * ldw + kbase + half * 16);
  v16bf r;
#pragma unroll
  for (int c = 0; c < 4; c++) {
    float4 f = p[c];
    cvt2(r, c * 4 + 0, f.x, f.y);
    cvt2(r, c * 4 + 2, f.z, f.w);
  }
  return r;
}

// B fragment (32x16) where memory is row-major [K][N] (e.g. V in attention).
__device__ __forceinline__ v16bf b_frag_rowmajor_f32(const float* Bm, int kbase,
                                                     long ldb, int nbase, int lane) {
  int n = lane & 15, half = (lane >> 4) & 1;
  v16bf r;
#pragma unroll
  for (int i = 0; i < 8; i++) {
    float x = Bm[(long)(kbase + half * 16 + 2 * i) * ldb + nbase + n];
    float y = Bm[(long)(kbase + half * 16 + 2 * i + 1) * ldb + nbase + n];
    cvt2(r, 2 * i, x, y);
  }
  return r;
}

// A fragment (16x32 bf16) from an LDS tile with row stride 32.
__device__ __forceinline__ v16bf a_frag_lds(const __bf16* L, int lane) {
  int m = lane & 15, half = (lane >> 4) & 1;
  v8bf a0 = *(const v8bf*)(L + m * 32 + half * 16);
  v8bf a1 = *(const v8bf*)(L + m * 32 + half * 16 + 8);
  v16bf r;
#pragma unroll
  for (int i = 0; i < 8; i++) { r[i] = a0[i]; r[8 + i] = a1[i]; }
  return r;
}

// ---------------------------------------------------------------------------
// Fused residual-add + RMSNorm: sum_out = a+b (fp32), norm_out = rmsnorm(a+b)*w (bf16)
// ---------------------------------------------------------------------------
__global__ void add_rmsnorm_kernel(const float* __restrict__ a,
                                   const float* __restrict__ b,
                                   const float* __restrict__ w,
                                   float* __restrict__ sum_out,
                                   __bf16* __restrict__ norm_out) {
  int t = blockIdx.x, tid = threadIdx.x;
  const float* ar = a + (long)t * DIM;
  const float* br = b + (long)t * DIM;
  float x[8]; float ss = 0.f;
#pragma unroll
  for (int j = 0; j < 8; j++) {
    int d = tid + j * 256;
    float v = ar[d] + br[d];
    x[j] = v; ss += v * v;
    sum_out[(long)t * DIM + d] = v;
  }
  __shared__ float red[256];
  red[tid] = ss; __syncthreads();
  for (int off = 128; off > 0; off >>= 1) {
    if (tid < off) red[tid] += red[tid + off];
    __syncthreads();
  }
  float rs = rsqrtf(red[0] / (float)DIM + 1e-6f);
#pragma unroll
  for (int j = 0; j < 8; j++) {
    int d = tid + j * 256;
    norm_out[(long)t * DIM + d] = f2bf(x[j] * rs * w[d]);
  }
}

// ---------------------------------------------------------------------------
// Dense GEMM: C[M,N] = A_bf16[M,K] @ W_f32[N,K]^T.
// Block = 4 waves; wave tile 32x64. Weight strip staged into LDS with
// global_load_async_to_lds_b128, double-buffered 2 k-steps ahead.
// Grid: (M/128, N/64)  (m fast -> the 4 m-blocks sharing a strip are adjacent).
// ---------------------------------------------------------------------------
__global__ void gemm_bf16_kernel(const __bf16* __restrict__ A,
                                 const float* __restrict__ W,
                                 float* __restrict__ C, int N, int K) {
  __shared__ float bsh[2][64 * 32];   // 2 x 8KB fp32 weight tiles
  int tid = threadIdx.x, lane = tid & 31, wave = tid >> 5;
  int mbase = blockIdx.x * 128 + wave * 32;
  int nbase = blockIdx.y * 64;
  int m = lane & 15, half = (lane >> 4) & 1;
  long arow0 = mbase + m, arow1 = mbase + 16 + m;

  // async staging map: 512 16B chunks (64 rows x 8), 4 per thread
  unsigned goff[4], loff[4];
#pragma unroll
  for (int is = 0; is < 4; is++) {
    int c = is * 128 + tid;
    int nrow = c >> 3, sub = c & 7;
    goff[is] = (unsigned)(((long)(nbase + nrow) * K + sub * 4) * 4);
    loff[is] = (unsigned)((nrow * 32 + sub * 4) * 4);
  }
  unsigned lb[2] = { (unsigned)(unsigned long long)(&bsh[0][0]),
                     (unsigned)(unsigned long long)(&bsh[1][0]) };

  v8f acc[2][4];
#pragma unroll
  for (int g = 0; g < 2; g++)
#pragma unroll
    for (int s = 0; s < 4; s++)
#pragma unroll
      for (int i = 0; i < 8; i++) acc[g][s][i] = 0.f;

  // prologue: stage k-steps 0 and 1
#pragma unroll
  for (int is = 0; is < 4; is++) ASYNC_LD16(lb[0] + loff[is], goff[is], W);
#pragma unroll
  for (int is = 0; is < 4; is++) ASYNC_LD16(lb[1] + loff[is], goff[is] + 128u, W);

  int NK = K / 32;
  for (int ks = 0; ks < NK; ks++) {
    int k = ks * 32;
    if (ks + 1 < NK) { WAIT_ASYNC(4); } else { WAIT_ASYNC(0); }
    __syncthreads();                       // tile for step ks ready block-wide
    const float* bw = &bsh[ks & 1][0];
    v16bf af0 = a_frag_bf16(A, arow0, K, k, lane);
    v16bf af1 = a_frag_bf16(A, arow1, K, k, lane);
#pragma unroll
    for (int s = 0; s < 4; s++) {
      v16bf bf_ = b_frag_lds_f32(bw, s * 16, lane);
      acc[0][s] = wmma_bf16(af0, bf_, acc[0][s]);
      acc[1][s] = wmma_bf16(af1, bf_, acc[1][s]);
    }
    __syncthreads();                       // tile consumed block-wide
    if (ks + 2 < NK) {
      unsigned kbyte = (unsigned)((ks + 2) * 32 * 4);
      unsigned lbn = lb[ks & 1];
#pragma unroll
      for (int is = 0; is < 4; is++) ASYNC_LD16(lbn + loff[is], goff[is] + kbyte, W);
    }
  }
#pragma unroll
  for (int g = 0; g < 2; g++)
#pragma unroll
    for (int s = 0; s < 4; s++)
#pragma unroll
      for (int j = 0; j < 8; j++) {
        int row = mbase + g * 16 + j + half * 8;
        int col = nbase + s * 16 + m;
        C[(long)row * N + col] = acc[g][s][j];
      }
}

// ---------------------------------------------------------------------------
// Grouped (gathered) expert GEMM with the same async-LDS weight staging.
// Grid: (4, N/64, E). Rows gathered via lists[e][slot] = aid (= t*6+k);
// A row = aid/6 (stage1) or aid (stage2); C row = aid.
// ---------------------------------------------------------------------------
__global__ void moe_gemm_kernel(const __bf16* __restrict__ A,
                                const float* __restrict__ W,
                                float* __restrict__ C,
                                const int* __restrict__ counts,
                                const int* __restrict__ lists,
                                int N, int K, int ldc, long wstride, int adiv6) {
  __shared__ float bsh[2][64 * 32];
  int e = blockIdx.z;
  int cnt = counts[e];
  if ((int)(blockIdx.x * 128) >= cnt) return;   // block-uniform exit, pre-barrier
  int tid = threadIdx.x, lane = tid & 31, wave = tid >> 5;
  int mbase = blockIdx.x * 128 + wave * 32;
  int nbase = blockIdx.y * 64;
  int m = lane & 15, half = (lane >> 4) & 1;
  const int* lst = lists + e * TOK;
  long arow[2];
#pragma unroll
  for (int g = 0; g < 2; g++) {
    int slot = mbase + g * 16 + m;
    int sc = slot < cnt ? slot : cnt - 1;
    int aid = lst[sc];
    arow[g] = adiv6 ? (long)(aid / TOPK) : (long)aid;
  }
  const float* We = W + (long)e * wstride;

  unsigned goff[4], loff[4];
#pragma unroll
  for (int is = 0; is < 4; is++) {
    int c = is * 128 + tid;
    int nrow = c >> 3, sub = c & 7;
    goff[is] = (unsigned)(((long)(nbase + nrow) * K + sub * 4) * 4);
    loff[is] = (unsigned)((nrow * 32 + sub * 4) * 4);
  }
  unsigned lb[2] = { (unsigned)(unsigned long long)(&bsh[0][0]),
                     (unsigned)(unsigned long long)(&bsh[1][0]) };

  v8f acc[2][4];
#pragma unroll
  for (int g = 0; g < 2; g++)
#pragma unroll
    for (int s = 0; s < 4; s++)
#pragma unroll
      for (int i = 0; i < 8; i++) acc[g][s][i] = 0.f;

#pragma unroll
  for (int is = 0; is < 4; is++) ASYNC_LD16(lb[0] + loff[is], goff[is], We);
#pragma unroll
  for (int is = 0; is < 4; is++) ASYNC_LD16(lb[1] + loff[is], goff[is] + 128u, We);

  int NK = K / 32;
  for (int ks = 0; ks < NK; ks++) {
    int k = ks * 32;
    if (ks + 1 < NK) { WAIT_ASYNC(4); } else { WAIT_ASYNC(0); }
    __syncthreads();
    const float* bw = &bsh[ks & 1][0];
    v16bf af0 = a_frag_bf16(A, arow[0], K, k, lane);
    v16bf af1 = a_frag_bf16(A, arow[1], K, k, lane);
#pragma unroll
    for (int s = 0; s < 4; s++) {
      v16bf bf_ = b_frag_lds_f32(bw, s * 16, lane);
      acc[0][s] = wmma_bf16(af0, bf_, acc[0][s]);
      acc[1][s] = wmma_bf16(af1, bf_, acc[1][s]);
    }
    __syncthreads();
    if (ks + 2 < NK) {
      unsigned kbyte = (unsigned)((ks + 2) * 32 * 4);
      unsigned lbn = lb[ks & 1];
#pragma unroll
      for (int is = 0; is < 4; is++) ASYNC_LD16(lbn + loff[is], goff[is] + kbyte, We);
    }
  }
#pragma unroll
  for (int g = 0; g < 2; g++)
#pragma unroll
    for (int j = 0; j < 8; j++) {
      int srow = mbase + g * 16 + j + half * 8;
      if (srow < cnt) {
        int said = lst[srow];
#pragma unroll
        for (int s = 0; s < 4; s++)
          C[(long)said * ldc + nbase + s * 16 + m] = acc[g][s][j];
      }
    }
}

// ---------------------------------------------------------------------------
// RoPE (neox rotate-halves) applied in place to q and k slices of qkv.
// ---------------------------------------------------------------------------
__global__ void rope_kernel(float* __restrict__ qkv, const int* __restrict__ pos) {
  int t = blockIdx.x;
  int idx = blockIdx.y * 256 + threadIdx.x;   // 0..2047
  int head = idx >> 6;
  int j = idx & 63;
  float p = (float)pos[t];
  float freq = __powf(10000.f, -2.f * (float)j / (float)HDIM);
  float ang = p * freq;
  float c = __cosf(ang), s = __sinf(ang);
  float* base = qkv + (long)t * QKVN + head * HDIM;
  float x1 = base[j], x2 = base[j + 64];
  base[j]      = x1 * c - x2 * s;
  base[j + 64] = x2 * c + x1 * s;
}

// ---------------------------------------------------------------------------
// Flash attention. Grid: (32 q-tiles, 4 head-groups), block 128 (4 waves).
// ---------------------------------------------------------------------------
__global__ void attn_kernel(const float* __restrict__ qkv,
                            __bf16* __restrict__ ctx_out) {
  __shared__ __bf16 psh[4][16 * 32];
  int lane = threadIdx.x & 31, wave = threadIdx.x >> 5;
  int qt = blockIdx.x;                 // 0..31
  int h = blockIdx.y * 4 + wave;       // 0..15
  int n_l = lane & 15, half = (lane >> 4) & 1;
  const float* Qb = qkv + h * HDIM;
  const float* Kb = qkv + DIM + h * HDIM;
  const float* Vb = qkv + 2 * DIM + h * HDIM;
  const long ld = QKVN;

  v16bf qf[4];
#pragma unroll
  for (int ks = 0; ks < 4; ks++)
    qf[ks] = a_frag_f32(Qb, (long)qt * 16 + n_l, ld, ks * 32, lane);

  v8f ctx[8];
#pragma unroll
  for (int dt = 0; dt < 8; dt++)
#pragma unroll
    for (int i = 0; i < 8; i++) ctx[dt][i] = 0.f;
  float rm[8], ls[8];
#pragma unroll
  for (int j = 0; j < 8; j++) { rm[j] = -1e30f; ls[j] = 0.f; }

  int nblocks = (qt + 2) >> 1;         // ceil((qt+1)*16 / 32)
  for (int b = 0; b < nblocks; b++) {
    int kb = b * 32;
    v8f S[2];
#pragma unroll
    for (int sub = 0; sub < 2; sub++)
#pragma unroll
      for (int i = 0; i < 8; i++) S[sub][i] = 0.f;

#pragma unroll
    for (int sub = 0; sub < 2; sub++)
#pragma unroll
      for (int ks = 0; ks < 4; ks++) {
        v16bf kf = b_frag_f32(Kb, kb + sub * 16, ld, ks * 32, lane);
        S[sub] = wmma_bf16(qf[ks], kf, S[sub]);
      }

    // scale + causal mask
#pragma unroll
    for (int sub = 0; sub < 2; sub++)
#pragma unroll
      for (int j = 0; j < 8; j++) {
        int mg = qt * 16 + j + half * 8;
        int key = kb + sub * 16 + n_l;
        float v = S[sub][j] * ATTN_SCALE;
        S[sub][j] = (key <= mg) ? v : -1e30f;
      }

    // online softmax update (row reductions within each 16-lane half)
#pragma unroll
    for (int j = 0; j < 8; j++) {
      float tmax = fmaxf(S[0][j], S[1][j]);
#pragma unroll
      for (int xm = 1; xm < 16; xm <<= 1) tmax = fmaxf(tmax, __shfl_xor(tmax, xm, 32));
      float newm = fmaxf(rm[j], tmax);
      float sc = __expf(rm[j] - newm);
      float p0 = __expf(S[0][j] - newm);
      float p1 = __expf(S[1][j] - newm);
      S[0][j] = p0; S[1][j] = p1;
      float rowsum = p0 + p1;
#pragma unroll
      for (int xm = 1; xm < 16; xm <<= 1) rowsum += __shfl_xor(rowsum, xm, 32);
      ls[j] = ls[j] * sc + rowsum;
      rm[j] = newm;
#pragma unroll
      for (int dt = 0; dt < 8; dt++) ctx[dt][j] *= sc;
    }

    // relayout P (C-layout) -> LDS -> A-layout fragment
    __syncthreads();
#pragma unroll
    for (int j = 0; j < 8; j++) {
      int mrow = j + half * 8;
      psh[wave][mrow * 32 + n_l]      = f2bf(S[0][j]);
      psh[wave][mrow * 32 + 16 + n_l] = f2bf(S[1][j]);
    }
    __syncthreads();
    v16bf pf = a_frag_lds(&psh[wave][0], lane);

#pragma unroll
    for (int dt = 0; dt < 8; dt++) {
      v16bf vf = b_frag_rowmajor_f32(Vb, kb, ld, dt * 16, lane);
      ctx[dt] = wmma_bf16(pf, vf, ctx[dt]);
    }
  }

  // finalize and emit bf16 context for the O-projection GEMM
#pragma unroll
  for (int j = 0; j < 8; j++) {
    float inv = 1.f / ls[j];
    int row = qt * 16 + j + half * 8;
#pragma unroll
    for (int dt = 0; dt < 8; dt++) {
      int col = h * HDIM + dt * 16 + n_l;
      ctx_out[(long)row * DIM + col] = f2bf(ctx[dt][j] * inv);
    }
  }
}

// ---------------------------------------------------------------------------
// Softmax gate + top-6 + per-expert assignment lists. One thread per token.
// ---------------------------------------------------------------------------
__global__ void topk_kernel(const float* __restrict__ logits,
                            float* __restrict__ topw,
                            int* __restrict__ counts,
                            int* __restrict__ lists) {
  int t = blockIdx.x * blockDim.x + threadIdx.x;
  if (t >= TOK) return;
  const float* lg = logits + (long)t * NEXP;
  float mx = -1e30f;
  for (int e = 0; e < NEXP; e++) mx = fmaxf(mx, lg[e]);
  float se = 0.f;
  for (int e = 0; e < NEXP; e++) se += __expf(lg[e] - mx);
  float inv = 1.f / se;
  unsigned long long used = 0ull;
  for (int k = 0; k < TOPK; k++) {
    float best = -1.f; int bi = 0;
    for (int e = 0; e < NEXP; e++) {
      if ((used >> e) & 1ull) continue;
      float p = __expf(lg[e] - mx);
      if (p > best) { best = p; bi = e; }
    }
    used |= 1ull << bi;
    topw[t * TOPK + k] = best * inv;
    int slot = atomicAdd(&counts[bi], 1);
    lists[bi * TOK + slot] = t * TOPK + k;    // aid
  }
}

// ---------------------------------------------------------------------------
// SwiGLU: act[r,i] = silu(gu[r,i]) * gu[r,I+i], emitted as bf16 for next GEMM.
// ---------------------------------------------------------------------------
__global__ void swiglu_kernel(const float* __restrict__ gu,
                              __bf16* __restrict__ act, int I) {
  int r = blockIdx.x;
  int i = blockIdx.y * blockDim.x + threadIdx.x;
  if (i >= I) return;
  float g = gu[(long)r * (2 * I) + i];
  float u = gu[(long)r * (2 * I) + I + i];
  float s = g / (1.f + __expf(-g));
  act[(long)r * I + i] = f2bf(s * u);
}

// ---------------------------------------------------------------------------
// Combine: out[t] = shared[t] + sum_k topw[t,k] * eout[t*6+k]  (fixed k order)
// ---------------------------------------------------------------------------
__global__ void combine_kernel(const float* __restrict__ shared_out,
                               const float* __restrict__ eout,
                               const float* __restrict__ topw,
                               float* __restrict__ out) {
  int t = blockIdx.x;
  int d = blockIdx.y * 256 + threadIdx.x;
  float acc = shared_out[(long)t * DIM + d];
#pragma unroll
  for (int k = 0; k < TOPK; k++)
    acc += topw[t * TOPK + k] * eout[(long)(t * TOPK + k) * DIM + d];
  out[(long)t * DIM + d] = acc;
}

// ---------------------------------------------------------------------------
extern "C" void kernel_launch(void* const* d_in, const int* in_sizes, int n_in,
                              void* d_out, int out_size, void* d_ws, size_t ws_size,
                              hipStream_t stream) {
  const int* positions   = (const int*)d_in[0];
  const float* hidden    = (const float*)d_in[1];
  const float* residual  = (const float*)d_in[2];
  const float* ln1w      = (const float*)d_in[3];
  const float* ln2w      = (const float*)d_in[4];
  const float* wqkv      = (const float*)d_in[5];
  const float* wo        = (const float*)d_in[6];
  const float* gate_w    = (const float*)d_in[7];
  const float* w1        = (const float*)d_in[8];
  const float* w2        = (const float*)d_in[9];
  const float* sw1       = (const float*)d_in[10];
  const float* sw2       = (const float*)d_in[11];

  float* out = (float*)d_out;
  float* residual2 = out + (long)TOK * DIM;   // second tuple output

  char* ws = (char*)d_ws;
  auto alloc = [&](size_t bytes) -> char* {
    char* p = ws; ws += (bytes + 255) & ~(size_t)255; return p;
  };
  float*  res1    = (float*)alloc((size_t)TOK * DIM * 4);
  __bf16* h_bf    = (__bf16*)alloc((size_t)TOK * DIM * 2);
  float*  qkv     = (float*)alloc((size_t)TOK * QKVN * 4);
  __bf16* ctx_bf  = (__bf16*)alloc((size_t)TOK * DIM * 2);
  float*  oproj   = (float*)alloc((size_t)TOK * DIM * 4);
  __bf16* xt_bf   = (__bf16*)alloc((size_t)TOK * DIM * 2);
  float*  logits  = (float*)alloc((size_t)TOK * NEXP * 4);
  float*  topw    = (float*)alloc((size_t)TOK * TOPK * 4);
  int*    counts  = (int*)alloc((size_t)NEXP * 4);
  int*    lists   = (int*)alloc((size_t)NEXP * TOK * 4);
  float*  gus     = (float*)alloc((size_t)TOK * 2 * ISHARE * 4);
  __bf16* acts    = (__bf16*)alloc((size_t)TOK * ISHARE * 2);
  float*  shout   = (float*)alloc((size_t)TOK * DIM * 4);
  float*  h1      = (float*)alloc((size_t)TOK * TOPK * 2 * IMOE * 4);
  __bf16* act1    = (__bf16*)alloc((size_t)TOK * TOPK * IMOE * 2);
  float*  eout    = (float*)alloc((size_t)TOK * TOPK * DIM * 4);

  // 1) residual add + RMSNorm (ln1)
  add_rmsnorm_kernel<<<TOK, 256, 0, stream>>>(hidden, residual, ln1w, res1, h_bf);
  // 2) QKV projection
  gemm_bf16_kernel<<<dim3(TOK / 128, QKVN / 64), 128, 0, stream>>>(h_bf, wqkv, qkv, QKVN, DIM);
  // 3) RoPE on q, k
  rope_kernel<<<dim3(TOK, 8), 256, 0, stream>>>(qkv, positions);
  // 4) causal flash attention
  attn_kernel<<<dim3(32, 4), 128, 0, stream>>>(qkv, ctx_bf);
  // 5) O projection
  gemm_bf16_kernel<<<dim3(TOK / 128, DIM / 64), 128, 0, stream>>>(ctx_bf, wo, oproj, DIM, DIM);
  // 6) residual add + RMSNorm (ln2); residual2 goes straight to d_out
  add_rmsnorm_kernel<<<TOK, 256, 0, stream>>>(oproj, res1, ln2w, residual2, xt_bf);
  // 7) gate logits
  gemm_bf16_kernel<<<dim3(TOK / 128, NEXP / 64), 128, 0, stream>>>(xt_bf, gate_w, logits, NEXP, DIM);
  // 8) softmax + top-6 + expert assignment lists
  hipMemsetAsync(counts, 0, NEXP * sizeof(int), stream);
  topk_kernel<<<2, 256, 0, stream>>>(logits, topw, counts, lists);
  // 9) shared expert
  gemm_bf16_kernel<<<dim3(TOK / 128, 2 * ISHARE / 64), 128, 0, stream>>>(xt_bf, sw1, gus, 2 * ISHARE, DIM);
  swiglu_kernel<<<dim3(TOK, (ISHARE + 255) / 256), 256, 0, stream>>>(gus, acts, ISHARE);
  gemm_bf16_kernel<<<dim3(TOK / 128, DIM / 64), 128, 0, stream>>>(acts, sw2, shout, DIM, ISHARE);
  // 10) routed experts (gathered grouped GEMMs)
  moe_gemm_kernel<<<dim3(4, 2 * IMOE / 64, NEXP), 128, 0, stream>>>(
      xt_bf, w1, h1, counts, lists, 2 * IMOE, DIM, 2 * IMOE, (long)2 * IMOE * DIM, 1);
  swiglu_kernel<<<dim3(TOK * TOPK, (IMOE + 255) / 256), 256, 0, stream>>>(h1, act1, IMOE);
  moe_gemm_kernel<<<dim3(4, DIM / 64, NEXP), 128, 0, stream>>>(
      act1, w2, eout, counts, lists, DIM, IMOE, DIM, (long)DIM * IMOE, 0);
  // 11) combine shared + weighted expert outputs
  combine_kernel<<<dim3(TOK, DIM / 256), 256, 0, stream>>>(shout, eout, topw, out);
}
